// QuantumNATModel_65481071396680
// MI455X (gfx1250) — matrix operand order repaired and checked by reading, not compiled, not run on hardware
//
#include <hip/hip_runtime.h>

typedef __attribute__((ext_vector_type(16))) _Float16 v16h;
typedef __attribute__((ext_vector_type(8)))  _Float16 v8h;
typedef __attribute__((ext_vector_type(8)))  float    v8f;
typedef __attribute__((ext_vector_type(4)))  float    v4f;
typedef __attribute__((ext_vector_type(2)))  float    v2f;

#define BSZ 8192
#define HP_PIX (17 * 16)          // padded 17x16 image, channel-last x8
#define HP_SZ  (HP_PIX * 8)       // halfs per sample

static __device__ __forceinline__ v16h cat16(v8h lo, v8h hi) {
  v16h r;
#pragma unroll
  for (int t = 0; t < 8; ++t) { r[t] = lo[t]; r[t + 8] = hi[t]; }
  return r;
}

// ---------------- conv1 (1->8, 3x3 SAME) + relu + maxpool2 ------------------
// Emits zero-bordered, channel-last padded image h1p[s][y][x][ic] (17x16x8 f16)
// so conv2 can stage its LDS tile with straight v8h copies.
__global__ __launch_bounds__(256) void k_conv1(const float* __restrict__ x,
                                               const float* __restrict__ w,
                                               const float* __restrict__ b,
                                               _Float16* __restrict__ h1p) {
  __shared__ float xs[28 * 28];
  __shared__ float ws[72];
  __shared__ float bs[8];
  const int tid = threadIdx.x, s = blockIdx.x;
  const v8h z8h = {};
  for (int i = tid; i < 784; i += 256) xs[i] = x[s * 784 + i];
  if (tid < 72) ws[tid] = w[tid];
  if (tid < 8)  bs[tid] = b[tid];
  for (int i = tid; i < HP_PIX; i += 256) {       // zero the 76 border pixels
    const int y = i / 16, xx = i % 16;
    if (y == 0 || y > 14 || xx == 0 || xx == 15)
      *(v8h*)&h1p[(size_t)s * HP_SZ + i * 8] = z8h;
  }
  __syncthreads();
  if (tid < 196) {
    const int py = tid / 14, px = tid % 14;
    v8h o;
#pragma unroll
    for (int oc = 0; oc < 8; ++oc) {
      float m = -1e30f;
#pragma unroll
      for (int sy = 0; sy < 2; ++sy)
#pragma unroll
        for (int sx = 0; sx < 2; ++sx) {
          const int y0 = 2 * py + sy, x0 = 2 * px + sx;
          float acc = bs[oc];
#pragma unroll
          for (int dy = 0; dy < 3; ++dy)
#pragma unroll
            for (int dx = 0; dx < 3; ++dx) {
              const int yy = y0 + dy - 1, xx = x0 + dx - 1;
              if (yy >= 0 && yy < 28 && xx >= 0 && xx < 28)
                acc += ws[oc * 9 + dy * 3 + dx] * xs[yy * 28 + xx];
            }
          m = fmaxf(m, acc);
        }
      o[oc] = (_Float16)fmaxf(m, 0.f);
    }
    *(v8h*)&h1p[(size_t)s * HP_SZ + ((py + 1) * 16 + (px + 1)) * 8] = o;
  }
}

// ---------------- conv2 (8->16) implicit GEMM via WMMA f16, + relu + pool ---
// K reordered as k = (dy*3+dx)*8 + ic  ->  each 8-wide K-group of the A-layout
// is one contiguous v8h LDS load from the zero-padded channel-last image.
__global__ __launch_bounds__(256) void k_conv2(const _Float16* __restrict__ h1p,
                                               const float* __restrict__ w,
                                               const float* __restrict__ b,
                                               _Float16* __restrict__ flatc,
                                               float* __restrict__ a_in) {
  __shared__ _Float16 hp[HP_SZ];          // hp[y][x][ic], zero borders
  __shared__ _Float16 wT[16][96];
  __shared__ float    outp[208][16];      // padded to 13*16 rows: no store guards
  __shared__ float    bsh[16];
  const int tid = threadIdx.x, s = blockIdx.x;
  const v8h z8h = {};
  for (int i = tid; i < HP_PIX; i += 256)         // straight copy (borders incl.)
    *(v8h*)&hp[i * 8] = *(const v8h*)&h1p[(size_t)s * HP_SZ + i * 8];
  for (int i = tid; i < 16 * 96; i += 256) {      // weights, reordered K
    const int oc = i / 96, k = i % 96;
    _Float16 v = (_Float16)0.f;
    if (k < 72) {
      const int g = k >> 3, ic = k & 7;           // g = dy*3+dx
      v = (_Float16)w[oc * 72 + ic * 9 + g];
    }
    wT[oc][k] = v;
  }
  if (tid < 16) bsh[tid] = b[tid];
  __syncthreads();
  const int wave = tid >> 5, lane = tid & 31;
  const int m16 = lane & 15, ha = lane >> 4;
  const float bias = bsh[m16];                    // hoisted; per-column of D
  v16h Bv[3];                                     // hoisted weight operands
#pragma unroll
  for (int ks = 0; ks < 3; ++ks) {
    v8h blo = *(const v8h*)&wT[m16][ks * 32 + ha * 8];
    v8h bhi = *(const v8h*)&wT[m16][ks * 32 + 16 + ha * 8];
    Bv[ks] = cat16(blo, bhi);
  }
  for (int mt = wave; mt < 13; mt += 8) {         // wave-uniform loop
    const int p = mt * 16 + m16;
    const int py = p / 14, px = p % 14;           // invalid p reads pad rows
    v8f acc;
#pragma unroll
    for (int r = 0; r < 8; ++r) acc[r] = bias;    // bias folded into C operand
#pragma unroll
    for (int ks = 0; ks < 2; ++ks) {              // slots 0..7: all real groups
      const int glo = ks * 4 + ha, ghi = glo + 2;
      v8h alo = *(const v8h*)&hp[((py + glo / 3) * 16 + (px + glo % 3)) * 8];
      v8h ahi = *(const v8h*)&hp[((py + ghi / 3) * 16 + (px + ghi % 3)) * 8];
      acc = __builtin_amdgcn_wmma_f32_16x16x32_f16(false, cat16(alo, ahi), false,
                                                   Bv[ks], (short)0, acc, false, false);
    }
    {                                             // ks=2: only slot 8 (g=8) real
      v8h t8 = *(const v8h*)&hp[((py + 2) * 16 + (px + 2)) * 8];
      v8h alo;
#pragma unroll
      for (int e = 0; e < 8; ++e) alo[e] = (ha == 0) ? t8[e] : (_Float16)0.f;
      acc = __builtin_amdgcn_wmma_f32_16x16x32_f16(false, cat16(alo, z8h), false,
                                                   Bv[2], (short)0, acc, false, false);
    }
#pragma unroll
    for (int r = 0; r < 8; ++r)                   // unconditional: outp padded
      outp[mt * 16 + r + 8 * ha][m16] = acc[r];
  }
  __syncthreads();
  for (int i = tid; i < 784; i += 256) {          // relu + 2x2 maxpool -> flat
    const int oc = i / 49, pp = i % 49;
    const int py = pp / 7, px = pp % 7;
    const float m0 = fmaxf(outp[(2 * py) * 14 + 2 * px][oc], outp[(2 * py) * 14 + 2 * px + 1][oc]);
    const float m1 = fmaxf(outp[(2 * py + 1) * 14 + 2 * px][oc], outp[(2 * py + 1) * 14 + 2 * px + 1][oc]);
    const float v = fmaxf(fmaxf(m0, m1), 0.f);
    flatc[s * 800 + i] = (_Float16)v;
    if (i < 4) a_in[s * 4 + i] = v;
  }
  if (tid < 16) flatc[s * 800 + 784 + tid] = (_Float16)0.f;  // pad cols
}

// ---------------- q = 0.5 * a@rot, k = a@ent, aT (f16, with ones row) -------
__global__ __launch_bounds__(256) void k_qk(const float* __restrict__ a_in,
                                            const float* __restrict__ rot,
                                            const float* __restrict__ ent,
                                            float* __restrict__ qg,
                                            float* __restrict__ kg,
                                            _Float16* __restrict__ aTg) {
  const int j = blockIdx.x * 256 + threadIdx.x;
  if (j >= BSZ) return;
  float a[4];
#pragma unroll
  for (int e = 0; e < 4; ++e) a[e] = a_in[j * 4 + e];
#pragma unroll
  for (int c = 0; c < 4; ++c) {
    float q = 0.f, k = 0.f;
#pragma unroll
    for (int e = 0; e < 4; ++e) { q += a[e] * rot[e * 4 + c]; k += a[e] * ent[e * 4 + c]; }
    qg[j * 4 + c] = q * 0.5f;                     // fold 1/sqrt(ed)
    kg[j * 4 + c] = k;
  }
#pragma unroll
  for (int c = 0; c < 8; ++c)
    aTg[c * BSZ + j] = (c < 4) ? (_Float16)a[c] : ((c == 4) ? (_Float16)1.f : (_Float16)0.f);
}

// ---------------- fc1 weight transpose+pad to f16 [64][800] -----------------
__global__ __launch_bounds__(256) void k_prepw(const float* __restrict__ w1,
                                               _Float16* __restrict__ w1T) {
  const int i = blockIdx.x * 256 + threadIdx.x;
  if (i >= 64 * 800) return;
  const int oc = i / 800, k = i % 800;
  w1T[i] = (k < 788) ? (_Float16)w1[oc * 788 + k] : (_Float16)0.f;
}

// ---------------- flash attention over 8192x8192, head dim 4 ----------------
// Staging uses GLOBAL_LOAD_ASYNC_TO_LDS_B128 (ASYNCcnt-tracked DMA to LDS).
// WMMA1: S^T = K_tile(16x4) x Q^T(4x16)  (exact f32, 16x16x4)
// WMMA2: attn^T|rowsum = Aext^T(16x32) x P^T(32x16) accumulated in f32
__global__ __launch_bounds__(256) void k_attn(const float* __restrict__ qg,
                                              const float* __restrict__ kg,
                                              const _Float16* __restrict__ aTg,
                                              _Float16* __restrict__ flatc) {
  __shared__ float    kls[1024 * 4];
  __shared__ _Float16 als[8][1024];
  const int tid = threadIdx.x;
  const int wave = tid >> 5, lane = tid & 31;
  const int m16 = lane & 15, ha = lane >> 4;
  const int iBase = (blockIdx.x * 8 + wave) * 16;
  v2f qv = *(const v2f*)&qg[(iBase + m16) * 4 + 2 * ha];   // B-op: K = 2*ha + v
  v8f acc = {};
  const v8f zero8 = {};
  for (int j0 = 0; j0 < BSZ; j0 += 1024) {
    __syncthreads();
    for (int i = tid; i < 1024; i += 256) {
      const unsigned dst = (unsigned)(uintptr_t)&kls[i * 4];
      asm volatile("global_load_async_to_lds_b128 %0, %1, off"
                   :: "v"(dst), "v"(&kg[(j0 + i) * 4]) : "memory");
    }
    for (int i = tid; i < 1024; i += 256) {
      const int c = i >> 7, col = (i & 127) * 8;
      const unsigned dst = (unsigned)(uintptr_t)&als[c][col];
      asm volatile("global_load_async_to_lds_b128 %0, %1, off"
                   :: "v"(dst), "v"(&aTg[c * BSZ + j0 + col]) : "memory");
    }
    asm volatile("s_wait_asynccnt 0x0" ::: "memory");
    __syncthreads();
    for (int jc = 0; jc < 64; ++jc) {
      const int jl = jc * 16;
      v2f kv = *(const v2f*)&kls[(jl + m16) * 4 + 2 * ha]; // A-op: K = 2*ha + v
      v8f S = __builtin_amdgcn_wmma_f32_16x16x4_f32(false, kv, false, qv, (short)0, zero8, false, false);
      float e[8];
#pragma unroll
      for (int r = 0; r < 8; ++r) e[r] = __expf(S[r]);     // S^T[j=r+8*ha][i=m16]
      v16h P;                                              // B-op build: half-wave swap
#pragma unroll
      for (int t = 0; t < 8; ++t) {
        const float oth = __shfl_xor(e[t], 16, 32);
        P[t]     = (ha == 0) ? (_Float16)e[t] : (_Float16)0.f;
        P[t + 8] = (ha == 0) ? (_Float16)oth  : (_Float16)0.f;
      }
      v8h alo = *(const v8h*)&als[m16 & 7][jl];            // A-op row c = m16
      v8h ahi = *(const v8h*)&als[m16 & 7][jl + 8];
      v16h Aa;
#pragma unroll
      for (int t = 0; t < 8; ++t) { Aa[t] = (ha == 0) ? alo[t] : (_Float16)0.f; Aa[t + 8] = (ha == 0) ? ahi[t] : (_Float16)0.f; }
      acc = __builtin_amdgcn_wmma_f32_16x16x32_f16(false, Aa, false, P, (short)0, acc, false, false);
    }
  }
  if (lane < 16) {                                // rows c=0..4 live in lanes 0-15
    const float inv = 1.0f / acc[4];              // rowsum via ones-column
#pragma unroll
    for (int c = 0; c < 4; ++c)
      flatc[(iBase + m16) * 800 + 784 + c] = (_Float16)(acc[c] * inv);
  }
}

// ---------------- fc1 (788->64) + relu + fc2 (64->4), WMMA f16 --------------
__global__ __launch_bounds__(256) void k_fc(const _Float16* __restrict__ flatc,
                                            const _Float16* __restrict__ w1T,
                                            const float* __restrict__ b1,
                                            const float* __restrict__ w2,
                                            const float* __restrict__ b2,
                                            float* __restrict__ outpre) {
  __shared__ _Float16 h1s[8][16][64];
  const int tid = threadIdx.x;
  const int wave = tid >> 5, lane = tid & 31;
  const int m16 = lane & 15, ha = lane >> 4;
  const int iBase = (blockIdx.x * 8 + wave) * 16;
  const int irow = iBase + m16;
  v8f C0 = {}, C1 = {}, C2 = {}, C3 = {};
  for (int ks = 0; ks < 25; ++ks) {
    const int k0 = ks * 32;
    __builtin_prefetch(&flatc[irow * 800 + k0 + 128], 0, 1);
    v8h alo = *(const v8h*)&flatc[irow * 800 + k0 + ha * 8];
    v8h ahi = *(const v8h*)&flatc[irow * 800 + k0 + 16 + ha * 8];
    v16h A = cat16(alo, ahi);
#pragma unroll
    for (int nt = 0; nt < 4; ++nt) {
      const int oc = nt * 16 + m16;
      v8h blo = *(const v8h*)&w1T[oc * 800 + k0 + ha * 8];
      v8h bhi = *(const v8h*)&w1T[oc * 800 + k0 + 16 + ha * 8];
      v16h Bv = cat16(blo, bhi);
      v8f& C = (nt == 0 ? C0 : nt == 1 ? C1 : nt == 2 ? C2 : C3);
      C = __builtin_amdgcn_wmma_f32_16x16x32_f16(false, A, false, Bv, (short)0, C, false, false);
    }
  }
#pragma unroll
  for (int nt = 0; nt < 4; ++nt) {                // h1 = relu(C + b1) -> LDS
    v8f C = (nt == 0 ? C0 : nt == 1 ? C1 : nt == 2 ? C2 : C3);
    const int n = nt * 16 + m16;
    const float bias = b1[n];
#pragma unroll
    for (int r = 0; r < 8; ++r)
      h1s[wave][r + 8 * ha][n] = (_Float16)fmaxf(C[r] + bias, 0.f);
  }
  __syncthreads();
  v8f D = {};
#pragma unroll
  for (int ks2 = 0; ks2 < 2; ++ks2) {             // fc2: 16x64 @ 64x16(pad)
    v8h alo = *(const v8h*)&h1s[wave][m16][ks2 * 32 + ha * 16];
    v8h ahi = *(const v8h*)&h1s[wave][m16][ks2 * 32 + ha * 16 + 8];
    v16h A = cat16(alo, ahi);
    v16h Bv;
#pragma unroll
    for (int t = 0; t < 16; ++t) {
      const int k = ks2 * 32 + ha * 16 + t;
      Bv[t] = (m16 < 4) ? (_Float16)w2[m16 * 64 + k] : (_Float16)0.f;
    }
    D = __builtin_amdgcn_wmma_f32_16x16x32_f16(false, A, false, Bv, (short)0, D, false, false);
  }
  if (m16 < 4) {
    const float bias = b2[m16];
#pragma unroll
    for (int r = 0; r < 8; ++r)
      outpre[(iBase + r + 8 * ha) * 4 + m16] = D[r] + bias;
  }
}

// ---------------- batchnorm stats + apply -----------------------------------
__global__ __launch_bounds__(256) void k_bnstat(const float* __restrict__ outpre,
                                                float* __restrict__ stats) {
  __shared__ float red[256][8];
  const int tid = threadIdx.x;
  float s[4] = {0, 0, 0, 0}, ss[4] = {0, 0, 0, 0};
  for (int i = tid; i < BSZ; i += 256) {
    v4f o = *(const v4f*)&outpre[i * 4];
#pragma unroll
    for (int c = 0; c < 4; ++c) { s[c] += o[c]; ss[c] += o[c] * o[c]; }
  }
#pragma unroll
  for (int c = 0; c < 4; ++c) { red[tid][c] = s[c]; red[tid][4 + c] = ss[c]; }
  __syncthreads();
  for (int off = 128; off > 0; off >>= 1) {
    if (tid < off)
#pragma unroll
      for (int c = 0; c < 8; ++c) red[tid][c] += red[tid + off][c];
    __syncthreads();
  }
  if (tid < 4) {
    const float mean = red[0][tid] * (1.f / BSZ);
    const float var = red[0][4 + tid] * (1.f / BSZ) - mean * mean;
    stats[tid] = mean;
    stats[4 + tid] = rsqrtf(var + 1e-5f);
  }
}

__global__ __launch_bounds__(256) void k_bnapply(const float* __restrict__ outpre,
                                                 const float* __restrict__ stats,
                                                 const float* __restrict__ gamma,
                                                 const float* __restrict__ beta,
                                                 float* __restrict__ out) {
  const int i = blockIdx.x * 256 + threadIdx.x;
  if (i >= BSZ * 4) return;
  const int c = i & 3;
  out[i] = gamma[c] * (outpre[i] - stats[c]) * stats[4 + c] + beta[c];
}

extern "C" void kernel_launch(void* const* d_in, const int* in_sizes, int n_in,
                              void* d_out, int out_size, void* d_ws, size_t ws_size,
                              hipStream_t stream) {
  const float* x   = (const float*)d_in[0];
  const float* c1w = (const float*)d_in[1];
  const float* c1b = (const float*)d_in[2];
  const float* c2w = (const float*)d_in[3];
  const float* c2b = (const float*)d_in[4];
  const float* rot = (const float*)d_in[5];
  const float* ent = (const float*)d_in[6];
  const float* w1  = (const float*)d_in[7];
  const float* b1  = (const float*)d_in[8];
  const float* w2  = (const float*)d_in[9];
  const float* b2  = (const float*)d_in[10];
  const float* gam = (const float*)d_in[11];
  const float* bet = (const float*)d_in[12];
  float* out = (float*)d_out;

  char* ws = (char*)d_ws;
  size_t off = 0;
  auto carve = [&](size_t bytes) {
    void* p = ws + off;
    off = (off + bytes + 255) & ~size_t(255);
    return p;
  };
  _Float16* h1p    = (_Float16*)carve((size_t)BSZ * HP_SZ * 2);  // padded conv1 out
  _Float16* flatc  = (_Float16*)carve((size_t)BSZ * 800 * 2);    // padded concat f16
  float*    a_in   = (float*)carve((size_t)BSZ * 4 * 4);
  float*    qg     = (float*)carve((size_t)BSZ * 4 * 4);
  float*    kg     = (float*)carve((size_t)BSZ * 4 * 4);
  _Float16* aTg    = (_Float16*)carve((size_t)8 * BSZ * 2);
  _Float16* w1T    = (_Float16*)carve((size_t)64 * 800 * 2);
  float*    outpre = (float*)carve((size_t)BSZ * 4 * 4);
  float*    stats  = (float*)carve(64);
  (void)ws_size; (void)in_sizes; (void)n_in; (void)out_size;

  k_conv1<<<BSZ, 256, 0, stream>>>(x, c1w, c1b, h1p);
  k_conv2<<<BSZ, 256, 0, stream>>>(h1p, c2w, c2b, flatc, a_in);
  k_qk<<<BSZ / 256, 256, 0, stream>>>(a_in, rot, ent, qg, kg, aTg);
  k_prepw<<<(64 * 800 + 255) / 256, 256, 0, stream>>>(w1, w1T);
  k_attn<<<64, 256, 0, stream>>>(qg, kg, aTg, flatc);
  k_fc<<<64, 256, 0, stream>>>(flatc, w1T, b1, w2, b2, outpre);
  k_bnstat<<<1, 256, 0, stream>>>(outpre, stats);
  k_bnapply<<<(BSZ * 4 + 255) / 256, 256, 0, stream>>>(outpre, stats, gam, bet, out);
}
